// MoELayer_88931592831375
// MI455X (gfx1250) — compile-verified, matrix operand order checked by reading, not run on hardware
//
#include <hip/hip_runtime.h>
#include <hip/hip_bf16.h>

#define N_EXP   8
#define TOPK    2
#define D_MODEL 1024
#define HIDDEN  4096
#define T_TOK   4096
#define TILE_M  32              // token tile: two 16-row WMMA M-tiles
#define CHUNK   512             // hidden chunk held in LDS per iteration
#define NHB     (CHUNK / 16)    // 32 hidden 16-col blocks per chunk
#define NDB     (D_MODEL / 16)  // 64 output 16-col blocks
#define WAVES   8               // 256 threads / wave32

typedef __attribute__((ext_vector_type(16))) __bf16 v16bf;
typedef __attribute__((ext_vector_type(8)))  float  v8f;

// ---------------------------------------------------------------------------
// Kernel 1: noisy-top-k gate (eval mode: no noise). One wave32 per token.
// Produces per-expert gathered token lists + gate values, and f/P sums.
// ---------------------------------------------------------------------------
__global__ void moe_gate_kernel(const float* __restrict__ x,
                                const float* __restrict__ wgate,
                                int*   __restrict__ cnt,
                                float* __restrict__ fsum,
                                float* __restrict__ Psum,
                                int*   __restrict__ list,
                                float* __restrict__ lgate) {
    const int token = blockIdx.x * WAVES + (threadIdx.x >> 5);
    const int lane  = threadIdx.x & 31;
    if (token >= T_TOK) return;

    const float* xr = x + (size_t)token * D_MODEL;
    float p[N_EXP];
#pragma unroll
    for (int e = 0; e < N_EXP; ++e) p[e] = 0.f;

    for (int d = lane; d < D_MODEL; d += 32) {
        const float xv = xr[d];
#pragma unroll
        for (int e = 0; e < N_EXP; ++e) p[e] += xv * wgate[e * D_MODEL + d];
    }
#pragma unroll
    for (int e = 0; e < N_EXP; ++e) {
        for (int off = 16; off > 0; off >>= 1) p[e] += __shfl_down(p[e], off, 32);
    }

    if (lane == 0) {
        // top-2 (first index wins ties, matching jax.lax.top_k)
        int i0 = 0; float l0 = p[0];
        for (int e = 1; e < N_EXP; ++e) if (p[e] > l0) { l0 = p[e]; i0 = e; }
        int i1 = -1; float l1 = -3.4e38f;
        for (int e = 0; e < N_EXP; ++e) if (e != i0 && p[e] > l1) { l1 = p[e]; i1 = e; }

        // softmax over the two selected logits
        const float g0 = __builtin_amdgcn_rcpf(1.f + __expf(l1 - l0));
        const float g1 = 1.f - g0;

        const int p0 = atomicAdd(&cnt[i0], 1);
        list[i0 * T_TOK + p0]  = token;
        lgate[i0 * T_TOK + p0] = g0;
        const int p1 = atomicAdd(&cnt[i1], 1);
        list[i1 * T_TOK + p1]  = token;
        lgate[i1 * T_TOK + p1] = g1;

        atomicAdd(&fsum[i0], g0);
        atomicAdd(&fsum[i1], g1);

        // full softmax over all 8 logits for P
        float m = p[0];
        for (int e = 1; e < N_EXP; ++e) m = fmaxf(m, p[e]);
        float s = 0.f, ex[N_EXP];
        for (int e = 0; e < N_EXP; ++e) { ex[e] = __expf(p[e] - m); s += ex[e]; }
        const float inv = __builtin_amdgcn_rcpf(s);
        for (int e = 0; e < N_EXP; ++e) atomicAdd(&Psum[e], ex[e] * inv);
    }
}

// ---------------------------------------------------------------------------
// Kernel 2: finalize aux loss
// ---------------------------------------------------------------------------
__global__ void moe_aux_kernel(const float* __restrict__ fsum,
                               const float* __restrict__ Psum,
                               float* __restrict__ aux_out) {
    if (threadIdx.x == 0 && blockIdx.x == 0) {
        float s = 0.f;
        const float invT = 1.f / (float)T_TOK;
        for (int e = 0; e < N_EXP; ++e) s += (fsum[e] * invT) * (Psum[e] * invT);
        *aux_out = (float)N_EXP * s;
    }
}

// ---------------------------------------------------------------------------
// Kernel 3: SwiGLU expert FFN over gathered tokens, bf16 WMMA, f32 accum.
// grid = (token_tile, expert)  [tile-major so resident blocks share one
// expert's 50 MB weight set -> L2-resident re-reads]; block = 8 wave32.
// Each block: 32 tokens = two 16-row M-tiles sharing every B fragment.
// ---------------------------------------------------------------------------
__global__ void __launch_bounds__(256)
moe_ffn_kernel(const float* __restrict__ x,
               const float* __restrict__ wg,
               const float* __restrict__ wu,
               const float* __restrict__ wd,
               const int*   __restrict__ cnt,
               const int*   __restrict__ list,
               const float* __restrict__ lgate,
               float* __restrict__ out) {
    __shared__ int   tidx[TILE_M];
    __shared__ float gs[TILE_M];
    __shared__ __align__(32) __bf16 xs[TILE_M][D_MODEL]; // 64 KB
    __shared__ __align__(32) __bf16 hs[TILE_M][CHUNK];   // 32 KB

    const int e  = blockIdx.y;
    const int nt = cnt[e];
    const int t0 = blockIdx.x * TILE_M;
    if (t0 >= nt) return;                 // uniform per block

    const int tid  = threadIdx.x;
    const int lane = tid & 31;
    const int wv   = tid >> 5;

    if (tid < TILE_M) {
        const int pth = t0 + tid;
        if (pth < nt) { tidx[tid] = list[e * T_TOK + pth]; gs[tid] = lgate[e * T_TOK + pth]; }
        else          { tidx[tid] = list[e * T_TOK];       gs[tid] = 0.f; } // padded row, gate 0
    }
    __syncthreads();

    // stage routed x tile to LDS as bf16 (float4 global loads)
    for (int i = tid; i < TILE_M * (D_MODEL / 4); i += 256) {
        const int r  = i >> 8;            // D_MODEL/4 == 256
        const int c4 = i & 255;
        const float4 v = ((const float4*)(x + (size_t)tidx[r] * D_MODEL))[c4];
        xs[r][c4 * 4 + 0] = (__bf16)v.x;
        xs[r][c4 * 4 + 1] = (__bf16)v.y;
        xs[r][c4 * 4 + 2] = (__bf16)v.z;
        xs[r][c4 * 4 + 3] = (__bf16)v.w;
    }
    __syncthreads();

    const float* wge = wg + (size_t)e * HIDDEN * D_MODEL;
    const float* wue = wu + (size_t)e * HIDDEN * D_MODEL;
    const float* wde = wd + (size_t)e * D_MODEL * HIDDEN;

    // fragment mapping used here (wave32; contiguous K, same for A and B):
    //  A (16x32 bf16): lane&15 = row M, (lane>>4)*16 = K half, 16 contiguous bf16
    //  B (32x16 bf16): lane&15 = col N, (lane>>4)*16 = K half
    //  C/D (16x16 f32): VGPR r -> M = r + 8*(lane>>4), N = lane&15
    const int row   = lane & 15;
    const int khalf = (lane >> 4) * 16;
    const int m0    = (lane >> 4) * 8;

    v8f acc[NDB / WAVES][2];              // persistent down-proj accumulators
#pragma unroll
    for (int i = 0; i < NDB / WAVES; ++i)
#pragma unroll
        for (int mt = 0; mt < 2; ++mt)
#pragma unroll
            for (int j = 0; j < 8; ++j) acc[i][mt][j] = 0.f;

    for (int ch = 0; ch < HIDDEN / CHUNK; ++ch) {
        const int hchunk = ch * CHUNK;

        // ---- stage 1: hidden chunk = silu(x Wg^T) * (x Wu^T) ----
        for (int hb = wv; hb < NHB; hb += WAVES) {
            const int hbase = hchunk + hb * 16;
            v8f cg[2], cu[2];
#pragma unroll
            for (int mt = 0; mt < 2; ++mt)
#pragma unroll
                for (int j = 0; j < 8; ++j) { cg[mt][j] = 0.f; cu[mt][j] = 0.f; }

            const float* pg = wge + (size_t)(hbase + row) * D_MODEL + khalf;
            const float* pu = wue + (size_t)(hbase + row) * D_MODEL + khalf;
            for (int k0 = 0; k0 < D_MODEL; k0 += 32) {
                __builtin_prefetch(pg + k0 + 64, 0, 0);
                __builtin_prefetch(pu + k0 + 64, 0, 0);
                v16bf bg, bu;
#pragma unroll
                for (int j = 0; j < 16; ++j) {
                    bg[j] = (__bf16)pg[k0 + j];
                    bu[j] = (__bf16)pu[k0 + j];
                }
#pragma unroll
                for (int mt = 0; mt < 2; ++mt) {   // B fragment feeds 2 M-tiles
                    const v16bf a = *(const v16bf*)&xs[mt * 16 + row][k0 + khalf];
                    cg[mt] = __builtin_amdgcn_wmma_f32_16x16x32_bf16(
                                 false, a, false, bg, (short)0, cg[mt], false, false);
                    cu[mt] = __builtin_amdgcn_wmma_f32_16x16x32_bf16(
                                 false, a, false, bu, (short)0, cu[mt], false, false);
                }
            }
#pragma unroll
            for (int mt = 0; mt < 2; ++mt)
#pragma unroll
                for (int r = 0; r < 8; ++r) {
                    const float g = cg[mt][r], u = cu[mt][r];
                    const float sig = __builtin_amdgcn_rcpf(1.f + __expf(-g));
                    hs[mt * 16 + m0 + r][hb * 16 + row] = (__bf16)(g * sig * u);
                }
        }
        __syncthreads();   // hidden chunk visible to all waves

        // ---- stage 2: accumulate y += hidden_chunk * Wd^T ----
#pragma unroll
        for (int i = 0; i < NDB / WAVES; ++i) {
            const int dbase = (wv + i * WAVES) * 16;
            const float* pd = wde + (size_t)(dbase + row) * HIDDEN + hchunk + khalf;
            for (int h0 = 0; h0 < CHUNK; h0 += 32) {
                __builtin_prefetch(pd + h0 + 64, 0, 0);
                v16bf b;
#pragma unroll
                for (int j = 0; j < 16; ++j) b[j] = (__bf16)pd[h0 + j];
#pragma unroll
                for (int mt = 0; mt < 2; ++mt) {
                    const v16bf a = *(const v16bf*)&hs[mt * 16 + row][h0 + khalf];
                    acc[i][mt] = __builtin_amdgcn_wmma_f32_16x16x32_bf16(
                                     false, a, false, b, (short)0, acc[i][mt], false, false);
                }
            }
        }
        __syncthreads();   // protect hs before next chunk overwrites it
    }

    // gate-scaled scatter-add into output (each token hit by 2 experts)
#pragma unroll
    for (int i = 0; i < NDB / WAVES; ++i) {
        const int dbase = (wv + i * WAVES) * 16;
#pragma unroll
        for (int mt = 0; mt < 2; ++mt)
#pragma unroll
            for (int r = 0; r < 8; ++r) {
                const int m = mt * 16 + m0 + r;
                atomicAdd(&out[(size_t)tidx[m] * D_MODEL + dbase + row],
                          gs[m] * acc[i][mt][r]);
            }
    }
}

// ---------------------------------------------------------------------------
// Host launcher
// ---------------------------------------------------------------------------
extern "C" void kernel_launch(void* const* d_in, const int* in_sizes, int n_in,
                              void* d_out, int out_size, void* d_ws, size_t ws_size,
                              hipStream_t stream) {
    (void)in_sizes; (void)n_in; (void)out_size; (void)ws_size;

    const float* x     = (const float*)d_in[0];
    const float* wgate = (const float*)d_in[1];
    // d_in[2] = w_noise: unused in eval mode
    const float* wg    = (const float*)d_in[3];
    const float* wu    = (const float*)d_in[4];
    const float* wd    = (const float*)d_in[5];
    float* out = (float*)d_out;

    // workspace layout
    char*  ws    = (char*)d_ws;
    int*   cnt   = (int*)ws;                                      // 8 ints
    float* fsum  = (float*)(ws + 64);                             // 8 floats
    float* Psum  = (float*)(ws + 128);                            // 8 floats
    int*   list  = (int*)(ws + 256);                              // 8*4096 ints
    float* lgate = (float*)(ws + 256 + sizeof(int) * N_EXP * T_TOK);

    hipMemsetAsync(ws, 0, 256, stream);                               // counters/sums
    hipMemsetAsync(out, 0, (size_t)T_TOK * D_MODEL * sizeof(float), stream);

    moe_gate_kernel<<<T_TOK / WAVES, 256, 0, stream>>>(x, wgate, cnt, fsum, Psum,
                                                       list, lgate);
    moe_aux_kernel<<<1, 32, 0, stream>>>(fsum, Psum, out + (size_t)T_TOK * D_MODEL);

    dim3 grid(T_TOK / TILE_M, N_EXP);   // tile-major: resident blocks share expert
    moe_ffn_kernel<<<grid, 256, 0, stream>>>(x, wg, wu, wd, cnt, list, lgate, out);
}
